// UResNet_57346403336380
// MI455X (gfx1250) — matrix operand order repaired and checked by reading, not compiled, not run on hardware
//
#include <hip/hip_runtime.h>
#include <hip/hip_bf16.h>

// ---------------------------------------------------------------------------
// Types for WMMA fragments (gfx1250, wave32)
// ---------------------------------------------------------------------------
typedef __bf16 bf16_t;
typedef __attribute__((ext_vector_type(16))) __bf16 v16bf;
typedef __attribute__((ext_vector_type(8)))  float  v8f;

union Frag {
    v16bf v;
    uint4 q[2];
};

#define NSTR 5
static const int NPLh[NSTR] = {16, 32, 48, 64, 80};

// ---------------------------------------------------------------------------
// WMMA gather-GEMM:  out[m, :] (+)= sum_k  x[nbr[m,k], :] @ Wt[k]^T
//   x   : (n_src, CIN)  bf16, row-major
//   Wt  : (kvol, cout, CIN) bf16  (pre-transposed: B loads contiguous)
//   nbr : (n_out, kvol) int32, -1 => zero row.  nbr==nullptr => identity.
// CIN is a compile-time constant -> K loop fully unrolled, tail masks folded.
// NT  = N tiles per wave (1 or 2); A fragment gathered once, reused per tile.
// ---------------------------------------------------------------------------
template <int CIN, int NT>
__global__ __launch_bounds__(256)
void subconv_wmma(const bf16_t* __restrict__ x, const bf16_t* __restrict__ Wt,
                  const int* __restrict__ nbr, float* __restrict__ out,
                  int n_out, int cout, int kvol, int col_base, int accflag)
{
    constexpr int KCH = (CIN + 31) / 32;      // 32-wide bf16 K chunks
    const int lane  = threadIdx.x & 31;
    const int wave  = threadIdx.x >> 5;
    const int row0  = (blockIdx.x * 8 + wave) * 16;
    if (row0 >= n_out) return;                // uniform per-wave exit
    const int col0  = col_base + blockIdx.y * (NT * 16);
    const int rlane = lane & 15;
    const int half  = lane >> 4;
    const int m     = row0 + rlane;
    const bool mval = (m < n_out);
    const uint4 Z   = make_uint4(0u, 0u, 0u, 0u);

    v8f acc[NT];
#pragma unroll
    for (int t = 0; t < NT; ++t) acc[t] = (v8f){};
    if (accflag) {
#pragma unroll
        for (int t = 0; t < NT; ++t)
#pragma unroll
            for (int r = 0; r < 8; ++r) {
                int row = row0 + half * 8 + r;
                if (row < n_out)
                    acc[t][r] = out[(size_t)row * cout + col0 + t * 16 + rlane];
            }
    }

    for (int k = 0; k < kvol; ++k) {
        int src = mval ? (nbr ? nbr[(size_t)m * kvol + k] : m) : -1;
        const bool rv = (src >= 0);
        const bf16_t* xrow = x + (size_t)(rv ? src : 0) * CIN;

        // ---- gather A fragments once per k (16-bit A 16x32 layout) ----
        Frag a[KCH];
#pragma unroll
        for (int ic = 0; ic < KCH; ++ic) {
            const int g0 = ic * 32 + half * 8;   // elements 0..7
            const int g1 = g0 + 16;              // elements 8..15
            a[ic].q[0] = (rv && g0 < CIN) ? *(const uint4*)(xrow + g0) : Z;
            a[ic].q[1] = (rv && g1 < CIN) ? *(const uint4*)(xrow + g1) : Z;
        }

        // ---- per N tile: B fragments + WMMA ----
#pragma unroll
        for (int t = 0; t < NT; ++t) {
            const bf16_t* wcol =
                Wt + ((size_t)k * cout + col0 + t * 16 + rlane) * CIN;
#pragma unroll
            for (int ic = 0; ic < KCH; ++ic) {
                // B 32x16: lanes 0-15 hold K=kc..kc+7,kc+16..+23, lanes 16-31 K=kc+8..+15,kc+24..+31
                const int h0 = ic * 32 + half * 8;
                Frag b;
                b.q[0] = (h0      < CIN) ? *(const uint4*)(wcol + h0)      : Z;
                b.q[1] = (h0 + 16 < CIN) ? *(const uint4*)(wcol + h0 + 16) : Z;
                acc[t] = __builtin_amdgcn_wmma_f32_16x16x32_bf16(
                             false, a[ic].v, false, b.v, (short)0, acc[t],
                             false, false);
            }
        }
    }

#pragma unroll
    for (int t = 0; t < NT; ++t)
#pragma unroll
        for (int r = 0; r < 8; ++r) {
            int row = row0 + half * 8 + r;
            if (row < n_out)
                out[(size_t)row * cout + col0 + t * 16 + rlane] = acc[t][r];
        }
}

// host-side dispatch for one conv: pairs of N tiles (NT=2) + 16-col remainder
template <int CIN>
static void launch_conv(const bf16_t* xb, const bf16_t* Wt, const int* nb,
                        int kvol, float* op, long nn, int cout, int acc,
                        hipStream_t stream)
{
    unsigned gx = (unsigned)((nn + 127) / 128);
    int pairs = cout / 32, rem = cout % 32;
    if (pairs) {
        dim3 g(gx, (unsigned)pairs);
        subconv_wmma<CIN, 2><<<g, 256, 0, stream>>>(xb, Wt, nb, op, (int)nn,
                                                    cout, kvol, 0, acc);
    }
    if (rem) {
        dim3 g(gx, 1u);
        subconv_wmma<CIN, 1><<<g, 256, 0, stream>>>(xb, Wt, nb, op, (int)nn,
                                                    cout, kvol, pairs * 32, acc);
    }
}

// ---------------------------------------------------------------------------
// BatchNorm pass 1 — per-channel sum / sumsq via LDS atomics + global atomics
// ---------------------------------------------------------------------------
__global__ __launch_bounds__(256)
void bn_reduce(const float* __restrict__ x, float* __restrict__ stats,
               long n, int c)
{
    extern __shared__ float s[];     // 2*c floats
    for (int i = threadIdx.x; i < 2 * c; i += blockDim.x) s[i] = 0.f;
    __syncthreads();
    long tot = n * (long)c;
    for (long idx = (long)blockIdx.x * blockDim.x + threadIdx.x; idx < tot;
         idx += (long)gridDim.x * blockDim.x) {
        float v = x[idx];
        int ch = (int)(idx % c);
        atomicAdd(&s[ch], v);
        atomicAdd(&s[c + ch], v * v);
    }
    __syncthreads();
    for (int i = threadIdx.x; i < 2 * c; i += blockDim.x)
        atomicAdd(&stats[i], s[i]);
}

// BatchNorm pass 2 — normalize + ReLU (LEAK==0), cast to OutT (bf16 or f32)
template <typename OutT>
__global__ __launch_bounds__(256)
void bn_norm(const float* __restrict__ x, const float* __restrict__ stats,
             const float* __restrict__ g, const float* __restrict__ b,
             OutT* __restrict__ y, long n, int c)
{
    long idx = (long)blockIdx.x * blockDim.x + threadIdx.x;
    if (idx >= n * (long)c) return;
    int ch    = (int)(idx % c);
    float inv = 1.f / (float)n;
    float mu  = stats[ch] * inv;
    float var = stats[c + ch] * inv - mu * mu;
    float v   = (x[idx] - mu) * rsqrtf(var + 1e-4f) * g[ch] + b[ch];
    v = v > 0.f ? v : 0.f;
    y[idx] = (OutT)v;
}

// ---------------------------------------------------------------------------
// Input conv: cin = 1, feature is column 4 of point_cloud (stride 5)
// ---------------------------------------------------------------------------
__global__ __launch_bounds__(256)
void in_conv(const float* __restrict__ pc, const float* __restrict__ W,
             const int* __restrict__ nbr, float* __restrict__ out, int n)
{
    long idx = (long)blockIdx.x * blockDim.x + threadIdx.x;
    if (idx >= (long)n * 16) return;
    int i = (int)(idx >> 4), o = (int)(idx & 15);
    float acc = 0.f;
    for (int k = 0; k < 27; ++k) {
        int s = nbr[(size_t)i * 27 + k];
        if (s >= 0) acc += pc[(size_t)s * 5 + 4] * W[k * 16 + o];
    }
    out[idx] = acc;
}

// ---------------------------------------------------------------------------
// Deconv (upsample): out_cat[i, colofs+o] = x[par[i]] . W[off[i]][:, o]
// ---------------------------------------------------------------------------
__global__ __launch_bounds__(256)
void deconv(const bf16_t* __restrict__ h, const float* __restrict__ W,
            const int* __restrict__ par, const int* __restrict__ off,
            float* __restrict__ dst, int n, int cin, int cout, int ld, int colofs)
{
    long idx = (long)blockIdx.x * blockDim.x + threadIdx.x;
    if (idx >= (long)n * cout) return;
    int i = (int)(idx / cout), o = (int)(idx % cout);
    const float*  Wk = W + (size_t)off[i] * cin * cout;
    const bf16_t* g  = h + (size_t)par[i] * cin;
    float acc = 0.f;
    for (int j = 0; j < cin; ++j) acc += (float)g[j] * Wk[(size_t)j * cout + o];
    dst[(size_t)i * ld + colofs + o] = acc;
}

// copy fmap into first half of concat buffer
__global__ __launch_bounds__(256)
void copy_cols(const float* __restrict__ src, float* __restrict__ dst,
               long n, int c, int ld)
{
    long idx = (long)blockIdx.x * blockDim.x + threadIdx.x;
    if (idx >= n * (long)c) return;
    long r = idx / c; int o = (int)(idx % c);
    dst[r * ld + o] = src[idx];
}

__global__ __launch_bounds__(256)
void cvt_bf16(const float* __restrict__ s, bf16_t* __restrict__ d, long n)
{
    long idx = (long)blockIdx.x * blockDim.x + threadIdx.x;
    if (idx < n) d[idx] = (bf16_t)s[idx];
}

// (kvol,cin,cout) f32  ->  (kvol,cout,cin) bf16  (WMMA-friendly B layout)
__global__ __launch_bounds__(256)
void transpose_cvt(const float* __restrict__ src, bf16_t* __restrict__ dst,
                   int kvol, int cin, int cout)
{
    long tot = (long)kvol * cin * cout;
    long i = (long)blockIdx.x * blockDim.x + threadIdx.x;
    if (i >= tot) return;
    long per = (long)cin * cout;
    int k  = (int)(i / per);
    long r = i - (long)k * per;
    int kk = (int)(r / cout);
    int nn = (int)(r % cout);
    dst[((size_t)k * cout + nn) * cin + kk] = (bf16_t)src[i];
}

// ---------------------------------------------------------------------------
// Host orchestration
// ---------------------------------------------------------------------------
static inline long cdiv(long a, long b) { return (a + b - 1) / b; }

extern "C" void kernel_launch(void* const* d_in, const int* in_sizes, int n_in,
                              void* d_out, int out_size, void* d_ws, size_t ws_size,
                              hipStream_t stream)
{
    (void)n_in; (void)out_size; (void)ws_size;
    // ------- input index map (mirrors setup_inputs insertion order) -------
    int idx = 0;
    const int pc_i = idx++;                       // point_cloud
    const int inW_i = idx++;                      // in_W
    struct Blk { int bn1g, bn1b, W1, bn2g, bn2b, W2; };
    Blk enc[NSTR][2];
    int down_bng[4], down_bnb[4], down_W[4];
    for (int i = 0; i < NSTR; ++i) {
        for (int j = 0; j < 2; ++j) {
            enc[i][j].bn1g = idx++; enc[i][j].bn1b = idx++; enc[i][j].W1 = idx++;
            enc[i][j].bn2g = idx++; enc[i][j].bn2b = idx++; enc[i][j].W2 = idx++;
        }
        if (i < NSTR - 1) { down_bng[i] = idx++; down_bnb[i] = idx++; down_W[i] = idx++; }
    }
    int up_bng[4], up_bnb[4], up_W[4], nin_i[4];
    Blk dec[4][2];
    for (int l = 0; l < NSTR - 1; ++l) {
        up_bng[l] = idx++; up_bnb[l] = idx++; up_W[l] = idx++; nin_i[l] = idx++;
        for (int j = 0; j < 2; ++j) {
            dec[l][j].bn1g = idx++; dec[l][j].bn1b = idx++; dec[l][j].W1 = idx++;
            dec[l][j].bn2g = idx++; dec[l][j].bn2b = idx++; dec[l][j].W2 = idx++;
        }
    }
    const int outg_i = idx++, outb_i = idx++;
    int nbr_i[NSTR];
    for (int l = 0; l < NSTR; ++l) nbr_i[l] = idx++;
    int dwn_i[4], par_i[4], off_i[4];
    for (int l = 0; l < NSTR - 1; ++l) { dwn_i[l] = idx++; par_i[l] = idx++; off_i[l] = idx++; }

    long n[NSTR];
    for (int l = 0; l < NSTR; ++l) n[l] = in_sizes[nbr_i[l]] / 27;

    // ------- workspace bump allocator -------
    char* ws = (char*)d_ws;
    size_t wo = 0;
    auto alloc = [&](size_t bytes) -> void* {
        wo = (wo + 255) & ~(size_t)255;
        void* p = ws + wo;
        wo += bytes;
        return p;
    };

    size_t maxe = 0;
    for (int l = 0; l < NSTR; ++l) {
        size_t e = (size_t)n[l] * 2 * NPLh[l];
        if (e > maxe) maxe = e;
    }

    float*  fm[NSTR];
    for (int l = 0; l < NSTR; ++l) fm[l] = (float*)alloc((size_t)n[l] * NPLh[l] * 4);
    float*  xcat[4];  bf16_t* xcatb[4];  float* xdec[4];
    for (int l = 0; l < 4; ++l) {
        xcat[l]  = (float*) alloc((size_t)n[l] * 2 * NPLh[l] * 4);
        xcatb[l] = (bf16_t*)alloc((size_t)n[l] * 2 * NPLh[l] * 2);
        xdec[l]  = (float*) alloc((size_t)n[l] * NPLh[l] * 4);
    }
    float*  tmpf  = (float*) alloc(maxe * 4);
    bf16_t* hb1   = (bf16_t*)alloc(maxe * 2);
    bf16_t* hb2   = (bf16_t*)alloc(maxe * 2);
    float*  stats = (float*) alloc(512 * 4);

    // ------- convert + transpose conv weights to bf16 -------
    auto wcvt = [&](int src, int kvol, int cin, int cout) -> bf16_t* {
        bf16_t* dst = (bf16_t*)alloc((size_t)kvol * cin * cout * 2);
        long tot = (long)kvol * cin * cout;
        transpose_cvt<<<cdiv(tot, 256), 256, 0, stream>>>(
            (const float*)d_in[src], dst, kvol, cin, cout);
        return dst;
    };
    bf16_t *encW[NSTR][2][2], *downW[4], *ninW[4], *decW[4][2][2];
    for (int i = 0; i < NSTR; ++i) {
        int c = NPLh[i];
        for (int j = 0; j < 2; ++j) {
            encW[i][j][0] = wcvt(enc[i][j].W1, 27, c, c);
            encW[i][j][1] = wcvt(enc[i][j].W2, 27, c, c);
        }
        if (i < NSTR - 1) downW[i] = wcvt(down_W[i], 8, c, NPLh[i + 1]);
    }
    for (int l = 0; l < 4; ++l) {
        int c = NPLh[l];
        ninW[l]       = wcvt(nin_i[l], 1, 2 * c, c);
        decW[l][0][0] = wcvt(dec[l][0].W1, 27, 2 * c, c);
        decW[l][0][1] = wcvt(dec[l][0].W2, 27, c, c);
        decW[l][1][0] = wcvt(dec[l][1].W1, 27, c, c);
        decW[l][1][1] = wcvt(dec[l][1].W2, 27, c, c);
    }

    // ------- helper launchers -------
    auto bnrelu_bf16 = [&](const float* xs, long nn, int cc, int gi, int bi, bf16_t* y) {
        hipMemsetAsync(stats, 0, 2 * cc * sizeof(float), stream);
        long tot = nn * (long)cc;
        int blocks = (int)cdiv(tot, 256); if (blocks > 2048) blocks = 2048;
        bn_reduce<<<blocks, 256, 2 * cc * sizeof(float), stream>>>(xs, stats, nn, cc);
        bn_norm<bf16_t><<<cdiv(tot, 256), 256, 0, stream>>>(
            xs, stats, (const float*)d_in[gi], (const float*)d_in[bi], y, nn, cc);
    };
    auto conv = [&](const bf16_t* xb, const bf16_t* Wt, const int* nb, int kvol,
                    float* op, long nn, int cin, int cout, int acc) {
        switch (cin) {
        case 16:  launch_conv<16 >(xb, Wt, nb, kvol, op, nn, cout, acc, stream); break;
        case 32:  launch_conv<32 >(xb, Wt, nb, kvol, op, nn, cout, acc, stream); break;
        case 48:  launch_conv<48 >(xb, Wt, nb, kvol, op, nn, cout, acc, stream); break;
        case 64:  launch_conv<64 >(xb, Wt, nb, kvol, op, nn, cout, acc, stream); break;
        case 80:  launch_conv<80 >(xb, Wt, nb, kvol, op, nn, cout, acc, stream); break;
        case 96:  launch_conv<96 >(xb, Wt, nb, kvol, op, nn, cout, acc, stream); break;
        case 128: launch_conv<128>(xb, Wt, nb, kvol, op, nn, cout, acc, stream); break;
        default: break;
        }
    };

    // ------- encoder -------
    in_conv<<<cdiv(n[0] * 16, 256), 256, 0, stream>>>(
        (const float*)d_in[pc_i], (const float*)d_in[inW_i],
        (const int*)d_in[nbr_i[0]], fm[0], (int)n[0]);

    for (int i = 0; i < NSTR; ++i) {
        int c = NPLh[i];
        const int* nb = (const int*)d_in[nbr_i[i]];
        float* x = fm[i];
        for (int j = 0; j < 2; ++j) {
            bnrelu_bf16(x, n[i], c, enc[i][j].bn1g, enc[i][j].bn1b, hb1);
            conv(hb1, encW[i][j][0], nb, 27, tmpf, n[i], c, c, 0);
            bnrelu_bf16(tmpf, n[i], c, enc[i][j].bn2g, enc[i][j].bn2b, hb2);
            conv(hb2, encW[i][j][1], nb, 27, x, n[i], c, c, 1);   // x += conv2
        }
        if (i < NSTR - 1) {
            bnrelu_bf16(x, n[i], c, down_bng[i], down_bnb[i], hb1);
            conv(hb1, downW[i], (const int*)d_in[dwn_i[i]], 8,
                 fm[i + 1], n[i + 1], c, NPLh[i + 1], 0);
        }
    }

    // ------- decoder -------
    for (int l = NSTR - 2; l >= 0; --l) {
        int c = NPLh[l], cu = NPLh[l + 1];
        const int* nb = (const int*)d_in[nbr_i[l]];
        const float* xin = (l == NSTR - 2) ? fm[NSTR - 1] : xdec[l + 1];
        long nu = n[l + 1];

        bnrelu_bf16(xin, nu, cu, up_bng[l], up_bnb[l], hb1);
        deconv<<<cdiv(n[l] * c, 256), 256, 0, stream>>>(
            hb1, (const float*)d_in[up_W[l]],
            (const int*)d_in[par_i[l]], (const int*)d_in[off_i[l]],
            xcat[l], (int)n[l], cu, c, 2 * c, c);
        copy_cols<<<cdiv(n[l] * c, 256), 256, 0, stream>>>(fm[l], xcat[l], n[l], c, 2 * c);
        cvt_bf16<<<cdiv(n[l] * 2 * c, 256), 256, 0, stream>>>(xcat[l], xcatb[l], n[l] * 2 * c);

        // dec block 0 (with nin skip projection)
        conv(xcatb[l], ninW[l], nullptr, 1, xdec[l], n[l], 2 * c, c, 0);   // sc = xcat @ nin
        bnrelu_bf16(xcat[l], n[l], 2 * c, dec[l][0].bn1g, dec[l][0].bn1b, hb1);
        conv(hb1, decW[l][0][0], nb, 27, tmpf, n[l], 2 * c, c, 0);
        bnrelu_bf16(tmpf, n[l], c, dec[l][0].bn2g, dec[l][0].bn2b, hb2);
        conv(hb2, decW[l][0][1], nb, 27, xdec[l], n[l], c, c, 1);

        // dec block 1
        bnrelu_bf16(xdec[l], n[l], c, dec[l][1].bn1g, dec[l][1].bn1b, hb1);
        conv(hb1, decW[l][1][0], nb, 27, tmpf, n[l], c, c, 0);
        bnrelu_bf16(tmpf, n[l], c, dec[l][1].bn2g, dec[l][1].bn2b, hb2);
        conv(hb2, decW[l][1][1], nb, 27, xdec[l], n[l], c, c, 1);
    }

    // ------- output BN (f32, leak 0) -------
    {
        long nn = n[0]; int cc = NPLh[0];
        hipMemsetAsync(stats, 0, 2 * cc * sizeof(float), stream);
        long tot = nn * (long)cc;
        int blocks = (int)cdiv(tot, 256); if (blocks > 2048) blocks = 2048;
        bn_reduce<<<blocks, 256, 2 * cc * sizeof(float), stream>>>(xdec[0], stats, nn, cc);
        bn_norm<float><<<cdiv(tot, 256), 256, 0, stream>>>(
            xdec[0], stats, (const float*)d_in[outg_i], (const float*)d_in[outb_i],
            (float*)d_out, nn, cc);
    }
}
